// LDGuidedRetention_41944650613077
// MI455X (gfx1250) — matrix operand order repaired and checked by reading, not compiled, not run on hardware
//
#include <hip/hip_runtime.h>

// LDGuidedRetention on gfx1250 (MI455X). All GEMMs via v_wmma_f32_16x16x32_f16.
// B=8, L=2048, D=768.
//   prep:      x -> f16; W_qkv, W_proj -> transposed f16
//   qkv_gemm:  qkv = x@Wqkv + b  -> q16 [B,L,D], k16 [B,L,D], vT16 [B,D,L]
//   attention: per (b, 16-row query tile): Q tile into LDS via TDM
//              (tensor_load_to_lds, padded LDS rows -> bank-conflict-free A
//              loads), 16x2048 score strip in LDS via WMMA, decay*1/sqrt(D),
//              row softmax (zeros above diagonal participate, as reference),
//              unnormalized P (f16) in LDS, O = P@V via WMMA, 1/sum folded
//              into epilogue -> o16.
//   proj_gemm: out = o16 @ WprojT + b_proj (f32).
// Pipelining strategy (tuned vs. round 2 which spilled): batch all 6 B-tile
// loads per K-chunk up front (progressive partial waits as the 6 WMMAs drain
// them) and double-buffer only the A tile (~112 vector regs, no scratch).

#define B_  8
#define L_  2048
#define D_  768
#define TD_ 2304  // 3*D

// LDS layout for attention (dynamic shared):
//   Qs: 16 rows, padded stride 1584 B (768 halfs + 16B pad per 512B -> banks 12*l)
//   Ss: 16 x 2052 f32  (stride 8208 B -> banks 4*m)
//   Ps: 16 x 2056 f16  (stride 4112 B -> banks 4*l)
#define QS_ROW_BYTES 1584
#define SS_STRIDE    2052
#define PS_STRIDE    2056
#define QS_BYTES     (16 * QS_ROW_BYTES)                 // 25344
#define SS_BYTES     (16 * SS_STRIDE * 4)                // 131328
#define PS_BYTES     (16 * PS_STRIDE * 2)                // 65792
#define LDS_TOTAL    (QS_BYTES + SS_BYTES + PS_BYTES + 64)

typedef __attribute__((ext_vector_type(16))) _Float16 v16h;
typedef __attribute__((ext_vector_type(8)))  float    v8f;
typedef __attribute__((ext_vector_type(4)))  unsigned int u32x4;
typedef __attribute__((ext_vector_type(8)))  int      i32x8;
typedef __attribute__((ext_vector_type(4)))  int      i32x4;

union V16HU { v16h h; uint4 q[2]; };

// A operand (16x32 f16, MxK): lane<16 -> row=lane,   K c+0..15
//                             lane>=16 -> row=lane-16, K c+16..31
static __device__ __forceinline__ v16h ld_a16(const _Float16* rowp, int dOff, int hs) {
  V16HU u;
  u.q[0] = *(const uint4*)(rowp + dOff + hs * 16);
  u.q[1] = *(const uint4*)(rowp + dOff + 8 + hs * 16);
  return u.h;
}

// B operand (32x16 f16, KxN): lane<16 -> col=lane, K 0..15; lane>=16 -> K 16..31
// => one contiguous 32B chunk per lane (caller applies + hs*16 halfs).
static __device__ __forceinline__ v16h ld_b16(const _Float16* p) {
  V16HU u;
  const uint4* p4 = (const uint4*)p;
  u.q[0] = p4[0];
  u.q[1] = p4[1];
  return u.h;
}

// A load from the padded Q tile: 16B pad inserted after every 512B of row data
// (matches TDM pad_interval=512B / pad_amount=16B). padded(off) = off + 16*(off/512).
static __device__ __forceinline__ v16h qs_ld_a(const char* Qs, int row, int dOff, int hs) {
  int o1 = dOff * 2 + hs * 32;
  int o2 = o1 + 16;
  V16HU u;
  u.q[0] = *(const uint4*)(Qs + row * QS_ROW_BYTES + o1 + ((o1 >> 9) << 4));
  u.q[1] = *(const uint4*)(Qs + row * QS_ROW_BYTES + o2 + ((o2 >> 9) << 4));
  return u.h;
}

static __device__ __forceinline__ v8f vzero8() {
  v8f z = {0.f, 0.f, 0.f, 0.f, 0.f, 0.f, 0.f, 0.f};
  return z;
}

static __device__ __forceinline__ v8f wmma_f16(v16h a, v16h b, v8f c) {
  return __builtin_amdgcn_wmma_f32_16x16x32_f16(false, a, false, b, (short)0, c,
                                                false, false);
}

// ---------------- prep kernels ----------------

__global__ void prep_convert_x(const float* __restrict__ x,
                               _Float16* __restrict__ x16, int n) {
  int i = blockIdx.x * blockDim.x + threadIdx.x;
  if (i < n) x16[i] = (_Float16)x[i];
}

// w: [rows][cols] f32  ->  wt: [cols][rows] f16
__global__ void prep_transpose_w(const float* __restrict__ w,
                                 _Float16* __restrict__ wt, int rows, int cols) {
  int i = blockIdx.x * blockDim.x + threadIdx.x;
  if (i < rows * cols) {
    int r = i / cols, c = i % cols;
    wt[(size_t)c * rows + r] = (_Float16)w[i];
  }
}

// ---------------- qkv projection ----------------
// Each wave computes a 16x96 strip (6 N-tiles).
__global__ void __launch_bounds__(256, 1)
qkv_gemm(const _Float16* __restrict__ x16,
         const _Float16* __restrict__ wqkvT,  // [2304][768]
         const float* __restrict__ b_qkv,
         _Float16* __restrict__ q16,
         _Float16* __restrict__ k16,
         _Float16* __restrict__ vT16) {
  const int wave = threadIdx.x >> 5;
  const int lane = threadIdx.x & 31;
  const int gw = blockIdx.x * 8 + wave;
  const int stripsN = TD_ / 96;      // 24
  const int tm = gw / stripsN;
  const int sn = gw % stripsN;
  const int hs = lane >> 4;
  const int ln = lane & 15;

  const _Float16* arow = x16 + (size_t)(tm * 16 + ln) * D_;
  const _Float16* bbase = wqkvT + (size_t)(sn * 96 + ln) * D_ + hs * 16;

  v8f acc[6];
#pragma unroll
  for (int t = 0; t < 6; ++t) acc[t] = vzero8();

  v16h a = ld_a16(arow, 0, hs);
  for (int dc = 0; dc < D_ / 32 - 1; ++dc) {
    v16h bt[6];
#pragma unroll
    for (int t = 0; t < 6; ++t)
      bt[t] = ld_b16(bbase + (size_t)t * 16 * D_ + dc * 32);
    v16h a_n = ld_a16(arow, (dc + 1) * 32, hs);  // prefetch next A
#pragma unroll
    for (int t = 0; t < 6; ++t) acc[t] = wmma_f16(a, bt[t], acc[t]);
    a = a_n;
  }
  {
    v16h bt[6];
#pragma unroll
    for (int t = 0; t < 6; ++t)
      bt[t] = ld_b16(bbase + (size_t)t * 16 * D_ + (D_ - 32));
#pragma unroll
    for (int t = 0; t < 6; ++t) acc[t] = wmma_f16(a, bt[t], acc[t]);
  }

#pragma unroll
  for (int t = 0; t < 6; ++t) {
    int col = sn * 96 + t * 16 + ln;
    float bias = b_qkv[col];
#pragma unroll
    for (int r = 0; r < 8; ++r) {
      int m = r + 8 * hs;
      int row = tm * 16 + m;
      _Float16 hv = (_Float16)(acc[t][r] + bias);
      if (col < D_) {
        q16[(size_t)row * D_ + col] = hv;
      } else if (col < 2 * D_) {
        k16[(size_t)row * D_ + (col - D_)] = hv;
      } else {
        int bb = row >> 11;
        int l  = row & (L_ - 1);
        vT16[((size_t)bb * D_ + (col - 2 * D_)) * L_ + l] = hv;  // V transposed
      }
    }
  }
}

// ---------------- attention ----------------
__global__ void __launch_bounds__(256, 1)
attention(const _Float16* __restrict__ q16,
          const _Float16* __restrict__ k16,
          const _Float16* __restrict__ vT16,
          const float* __restrict__ gammaPtr,
          _Float16* __restrict__ o16) {
  extern __shared__ char smem[];
  char*     Qs   = smem;                               // padded Q tile
  float*    Ss   = (float*)(smem + QS_BYTES);          // 16 x SS_STRIDE f32
  _Float16* Ps   = (_Float16*)(smem + QS_BYTES + SS_BYTES);  // 16 x PS_STRIDE f16
  float*    Linv = (float*)(smem + QS_BYTES + SS_BYTES + PS_BYTES);

  const int b     = blockIdx.y;
  const int qt    = blockIdx.x;
  const int tid   = threadIdx.x;
  const int wave  = tid >> 5;
  const int lane  = tid & 31;
  const int hs    = lane >> 4;
  const int ln    = lane & 15;
  const int qbase = qt * 16;

  const float gamma = gammaPtr[0];
  const float lg    = __logf(gamma);
  const float scale = 0.03608439182435161f;  // 1/sqrt(768)

  const _Float16* qtile = q16 + ((size_t)b * L_ + qbase) * D_;

#if __has_builtin(__builtin_amdgcn_tensor_load_to_lds) && __has_builtin(__builtin_amdgcn_s_wait_tensorcnt)
  // Tensor Data Mover: DMA the 16x768 f16 Q tile into LDS with padding
  // (pad_interval = 512B -> code 6, pad_amount = 16B -> code 3) so A-operand
  // ds_load_b128 hit distinct banks. One TDM op per block, issued by wave 0.
  if (wave == 0) {
    unsigned long long ga = (unsigned long long)(size_t)qtile;
    unsigned ldsA = (unsigned)(size_t)(void*)Qs;
    u32x4 g0;
    g0[0] = 1u;                                   // count=1
    g0[1] = ldsA;                                 // lds_addr
    g0[2] = (unsigned)ga;                         // global_addr[31:0]
    g0[3] = (unsigned)((ga >> 32) & 0x1FFFFFFu) | 0x80000000u;  // addr[56:32] | type=2
    i32x8 g1;
    g1[0] = (int)((1u << 16) |                    // data_size = 2 bytes
                  (1u << 20) |                    // pad_enable
                  (6u << 22) |                    // pad_interval: 128 dwords (512B)
                  (3u << 25));                    // pad_amount: 4 dwords (16B)
    g1[1] = (int)(768u << 16);                    // tensor_dim0 = 768
    g1[2] = (int)(16u << 16);                     // tensor_dim1 = 16
    g1[3] = (int)(768u << 16);                    // tile_dim0 = 768
    g1[4] = (int)16;                              // tile_dim1 = 16
    g1[5] = (int)768;                             // tensor_dim0_stride = 768
    g1[6] = 0;
    g1[7] = 0;
    i32x4 gz4;
    gz4[0] = 0; gz4[1] = 0; gz4[2] = 0; gz4[3] = 0;
    i32x8 gz8;
    gz8[0] = 0; gz8[1] = 0; gz8[2] = 0; gz8[3] = 0;
    gz8[4] = 0; gz8[5] = 0; gz8[6] = 0; gz8[7] = 0;
    // 6-arg form (clang-23 / therock-10 headers): groups 0..3 + extra group + cpol.
    __builtin_amdgcn_tensor_load_to_lds(g0, g1, gz4, gz4, gz8, 0);
    __builtin_amdgcn_s_wait_tensorcnt(0);
  }
#else
  // Fallback: manual copy with the same padded layout.
  {
    const uint4* src = (const uint4*)qtile;
    for (int i = tid; i < 16 * 96; i += 256) {  // 96 x 16B chunks per row
      int row = i / 96, c = i % 96;
      int o = c * 16;
      *(uint4*)(Qs + row * QS_ROW_BYTES + o + ((o >> 9) << 4)) = src[row * 96 + c];
    }
  }
#endif
  __syncthreads();

  // Phase 1: score strip. Wave w owns keys [w*256, w*256+256) -> 16 key tiles.
  for (int kt = 0; kt < 16; ++kt) {
    const int j0 = wave * 256 + kt * 16;
    if (j0 > qbase + 15) {
      // Fully above the diagonal -> masked score is exactly 0 (no WMMA needed).
#pragma unroll
      for (int r = 0; r < 8; ++r) Ss[(r + 8 * hs) * SS_STRIDE + j0 + ln] = 0.f;
      continue;
    }
    v8f acc = vzero8();
    const _Float16* krow = k16 + ((size_t)b * L_ + j0 + ln) * D_ + hs * 16;
    v16h a  = qs_ld_a(Qs, ln, 0, hs);
    v16h bm = ld_b16(krow);
    for (int dc = 0; dc < D_ / 32 - 1; ++dc) {
      v16h a_n = qs_ld_a(Qs, ln, (dc + 1) * 32, hs);
      v16h b_n = ld_b16(krow + (dc + 1) * 32);
      acc = wmma_f16(a, bm, acc);
      a = a_n;
      bm = b_n;
    }
    acc = wmma_f16(a, bm, acc);
#pragma unroll
    for (int r = 0; r < 8; ++r) {
      int m = r + 8 * hs;
      int i = qbase + m;
      int j = j0 + ln;
      float v = 0.f;
      if (j <= i) v = acc[r] * scale * __expf((float)(i - j) * lg);
      Ss[m * SS_STRIDE + j] = v;  // j>i stays 0 and still enters the softmax
    }
  }
  __syncthreads();

  // Phase 2: row softmax over the full 2048-wide row (zeros included).
  // Store unnormalized exp(s-m) as f16; 1/sum folded into the O epilogue.
  for (int rr = 0; rr < 2; ++rr) {
    int row = wave * 2 + rr;
    const float* srow = Ss + (size_t)row * SS_STRIDE;
    float mx = -3.0e38f;
    for (int j = lane; j < L_; j += 32) mx = fmaxf(mx, srow[j]);
#pragma unroll
    for (int off = 16; off > 0; off >>= 1) mx = fmaxf(mx, __shfl_xor(mx, off, 32));
    float sum = 0.f;
    _Float16* prow = Ps + (size_t)row * PS_STRIDE;
    for (int j = lane; j < L_; j += 32) {
      float p = __expf(srow[j] - mx);
      sum += p;
      prow[j] = (_Float16)p;
    }
#pragma unroll
    for (int off = 16; off > 0; off >>= 1) sum += __shfl_xor(sum, off, 32);
    if (lane == 0) Linv[row] = 1.0f / sum;
  }
  __syncthreads();

  // Phase 3: O = P @ V. Wave w owns output columns [w*96, w*96+96) (6 tiles).
  v8f acc[6];
#pragma unroll
  for (int t = 0; t < 6; ++t) acc[t] = vzero8();
  const _Float16* prow = Ps + (size_t)ln * PS_STRIDE;
  const _Float16* vbase = vT16 + ((size_t)b * D_ + wave * 96 + ln) * L_ + hs * 16;

  v16h a = ld_a16(prow, 0, hs);
  for (int kc = 0; kc < L_ / 32 - 1; ++kc) {
    v16h bt[6];
#pragma unroll
    for (int t = 0; t < 6; ++t)
      bt[t] = ld_b16(vbase + (size_t)t * 16 * L_ + kc * 32);
    v16h a_n = ld_a16(prow, (kc + 1) * 32, hs);  // prefetch next A
#pragma unroll
    for (int t = 0; t < 6; ++t) acc[t] = wmma_f16(a, bt[t], acc[t]);
    a = a_n;
  }
  {
    v16h bt[6];
#pragma unroll
    for (int t = 0; t < 6; ++t)
      bt[t] = ld_b16(vbase + (size_t)t * 16 * L_ + (L_ - 32));
#pragma unroll
    for (int t = 0; t < 6; ++t) acc[t] = wmma_f16(a, bt[t], acc[t]);
  }

#pragma unroll
  for (int t = 0; t < 6; ++t) {
    int col = wave * 96 + t * 16 + ln;
#pragma unroll
    for (int r = 0; r < 8; ++r) {
      int m = r + 8 * hs;
      float v = acc[t][r] * Linv[m];
      o16[((size_t)b * L_ + qbase + m) * D_ + col] = (_Float16)v;
    }
  }
}

// ---------------- output projection ----------------
__global__ void __launch_bounds__(256, 1)
proj_gemm(const _Float16* __restrict__ o16,
          const _Float16* __restrict__ wprojT,  // [768][768]
          const float* __restrict__ b_proj,
          float* __restrict__ out) {
  const int wave = threadIdx.x >> 5;
  const int lane = threadIdx.x & 31;
  const int gw = blockIdx.x * 8 + wave;
  const int stripsN = D_ / 96;       // 8
  const int tm = gw / stripsN;
  const int sn = gw % stripsN;
  const int hs = lane >> 4;
  const int ln = lane & 15;

  const _Float16* arow = o16 + (size_t)(tm * 16 + ln) * D_;
  const _Float16* bbase = wprojT + (size_t)(sn * 96 + ln) * D_ + hs * 16;

  v8f acc[6];
#pragma unroll
  for (int t = 0; t < 6; ++t) acc[t] = vzero8();

  v16h a = ld_a16(arow, 0, hs);
  for (int dc = 0; dc < D_ / 32 - 1; ++dc) {
    v16h bt[6];
#pragma unroll
    for (int t = 0; t < 6; ++t)
      bt[t] = ld_b16(bbase + (size_t)t * 16 * D_ + dc * 32);
    v16h a_n = ld_a16(arow, (dc + 1) * 32, hs);  // prefetch next A
#pragma unroll
    for (int t = 0; t < 6; ++t) acc[t] = wmma_f16(a, bt[t], acc[t]);
    a = a_n;
  }
  {
    v16h bt[6];
#pragma unroll
    for (int t = 0; t < 6; ++t)
      bt[t] = ld_b16(bbase + (size_t)t * 16 * D_ + (D_ - 32));
#pragma unroll
    for (int t = 0; t < 6; ++t) acc[t] = wmma_f16(a, bt[t], acc[t]);
  }

#pragma unroll
  for (int t = 0; t < 6; ++t) {
    int col = sn * 96 + t * 16 + ln;
    float bias = b_proj[col];
#pragma unroll
    for (int r = 0; r < 8; ++r) {
      int m = r + 8 * hs;
      out[(size_t)(tm * 16 + m) * D_ + col] = acc[t][r] + bias;
    }
  }
}

// ---------------- launch ----------------
extern "C" void kernel_launch(void* const* d_in, const int* in_sizes, int n_in,
                              void* d_out, int out_size, void* d_ws, size_t ws_size,
                              hipStream_t stream) {
  const float* x      = (const float*)d_in[0];
  const float* gamma  = (const float*)d_in[1];
  const float* W_qkv  = (const float*)d_in[2];
  const float* b_qkv  = (const float*)d_in[3];
  const float* W_proj = (const float*)d_in[4];
  const float* b_proj = (const float*)d_in[5];
  float* out = (float*)d_out;

  const size_t n_xd = (size_t)B_ * L_ * D_;  // 12,582,912
  _Float16* x16    = (_Float16*)d_ws;
  _Float16* q16    = x16 + n_xd;
  _Float16* k16    = q16 + n_xd;
  _Float16* vT16   = k16 + n_xd;
  _Float16* o16    = vT16 + n_xd;
  _Float16* wqkvT  = o16 + n_xd;
  _Float16* wprojT = wqkvT + (size_t)D_ * TD_;

  prep_convert_x<<<((int)n_xd + 255) / 256, 256, 0, stream>>>(x, x16, (int)n_xd);
  prep_transpose_w<<<(D_ * TD_ + 255) / 256, 256, 0, stream>>>(W_qkv, wqkvT, D_, TD_);
  prep_transpose_w<<<(D_ * D_ + 255) / 256, 256, 0, stream>>>(W_proj, wprojT, D_, D_);

  qkv_gemm<<<(B_ * L_ / 16) * (TD_ / 96) / 8, 256, 0, stream>>>(
      x16, wqkvT, b_qkv, q16, k16, vT16);

  dim3 ga(L_ / 16, B_);
  attention<<<ga, 256, (size_t)LDS_TOTAL, stream>>>(q16, k16, vT16, gamma, o16);

  proj_gemm<<<(B_ * L_ / 16) * (D_ / 96) / 8, 256, 0, stream>>>(
      o16, wprojT, b_proj, out);
}
